// FastSpatialWindowAttention_76295799046375
// MI455X (gfx1250) — matrix operand order repaired and checked by reading, not compiled, not run on hardware
//
#include <hip/hip_runtime.h>
#include <stdint.h>

// ---------------------------------------------------------------------------
// FastSpatialWindowAttention (Swin window attention), MI455X / gfx1250
// B=4, H=W=256, C=192, NH=6, hd=32, WS=8 -> 4096 windows of 64 tokens.
// fp32 -> bf16 on the fly, all GEMMs via v_wmma_f32_16x16x32_bf16 (f32 acc),
// fully fused per-window kernel. Wave-uniform ids forced scalar via
// readfirstlane; weight stream prefetched one tile ahead.
// ---------------------------------------------------------------------------

typedef __attribute__((ext_vector_type(16))) __bf16 v16bf;
typedef __attribute__((ext_vector_type(8)))  float  v8f;

#define CCH   192      // channels
#define TOK   64       // tokens per window (8x8)
#define NHEAD 6
#define HD    32
#define WPX   32       // windows per image row/col (256/8)
#define IMGW  256

__device__ __forceinline__ __bf16 f2bf(float f) {
    return (__bf16)f;   // native f32->bf16 convert on gfx1250
}

// A-matrix 16x32 bf16 fragment from row-major [16][stride] tile.
// lane<16: M=lane, K = {0..7, 16..23}; lane>=16: M=lane-16, K = {8..15, 24..31}
__device__ __forceinline__ v16bf load_A(const __bf16* base, int stride,
                                        int ln, int hi) {
    const __bf16* p = base + ln * stride + hi * 8;
    v16bf a;
#pragma unroll
    for (int j = 0; j < 8; ++j) a[j]     = p[j];
#pragma unroll
    for (int j = 0; j < 8; ++j) a[8 + j] = p[16 + j];
    return a;
}

// B-matrix 32x16 bf16 fragment; memory holds B as [N][K] row-major
// (weight row / key-token row contiguous in K).
// lane<16: N=lane, K = {0..7, 16..23}; lane>=16: N=lane-16, K = {8..15, 24..31}
__device__ __forceinline__ v16bf load_B(const __bf16* base, int stride,
                                        int ln, int hi) {
    const __bf16* p = base + ln * stride + hi * 8;
    v16bf b;
#pragma unroll
    for (int j = 0; j < 8; ++j) b[j]     = p[j];
#pragma unroll
    for (int j = 0; j < 8; ++j) b[8 + j] = p[16 + j];
    return b;
}

__device__ __forceinline__ v8f wmma_bf16(const v16bf& a, const v16bf& b,
                                         const v8f& c) {
    return __builtin_amdgcn_wmma_f32_16x16x32_bf16(
        /*neg_a=*/false, a, /*neg_b=*/false, b,
        /*c_mod=*/(short)0, c, /*reuse_a=*/false, /*reuse_b=*/false);
}

// ------------------------- weight conversion -------------------------------
__global__ void cvt_weights_kernel(const float* __restrict__ qkv_w,
                                   const float* __restrict__ proj_w,
                                   __bf16* __restrict__ qw,
                                   __bf16* __restrict__ pw) {
    int i = blockIdx.x * blockDim.x + threadIdx.x;
    if (i < 3 * CCH * CCH) qw[i] = f2bf(qkv_w[i]);
    if (i < CCH * CCH)     pw[i] = f2bf(proj_w[i]);
}

// ------------------------- fused window attention --------------------------
__global__ __launch_bounds__(256, 2)
void swin_window_attn_kernel(const float* __restrict__ x,
                             const __bf16* __restrict__ qw,   // [576][192] bf16
                             const __bf16* __restrict__ pw,   // [192][192] bf16
                             const float* __restrict__ proj_b,
                             const float* __restrict__ rpb,   // [225][6] f32
                             float* __restrict__ out) {
    extern __shared__ char smem[];
    __bf16* xo  = (__bf16*)smem;            // [64][192] x, later O  (24576 B)
    __bf16* qb  = xo + TOK * CCH;           // [64][192] Q           (24576 B)
    __bf16* kb  = qb + TOK * CCH;           // [64][192] K           (24576 B)
    __bf16* vT  = kb + TOK * CCH;           // [192][64] V^T         (24576 B)
    float*  Sb  = (float*)(vT + TOK * CCH); // [64][64] scores f32   (16384 B)
    __bf16* Pb  = (__bf16*)(Sb + TOK * TOK);// [64][64] probs bf16   ( 8192 B)
    float*  bia = (float*)(Pb + TOK * TOK); // [225][6] bias         ( 5400 B)

    const int tid  = threadIdx.x;
    const int lane = tid & 31;
    // wave id as SGPR: makes tile loops / QKV-store branches scalar control flow
    const int wv   = __builtin_amdgcn_readfirstlane(tid >> 5);   // 0..7
    const int ln   = lane & 15;
    const int hi   = lane >> 4;     // lane half 0/1
    const float scale = 0.17677669529663687f;   // 1/sqrt(32)

    const int wi = blockIdx.x;
    const int bb = wi >> 10;                 // batch
    const int wy = (wi >> 5) & 31;
    const int wx = wi & 31;
    const float* xb = x + (size_t)bb * IMGW * IMGW * CCH;

    // ---- stage in x window (f32 -> bf16) + bias table ----
    for (int e = tid; e < TOK * CCH; e += 256) {
        int t = e / CCH, ch = e - t * CCH;
        int row = wy * 8 + (t >> 3), col = wx * 8 + (t & 7);
        xo[e] = f2bf(xb[((size_t)row * IMGW + col) * CCH + ch]);
    }
    for (int e = tid; e < 225 * NHEAD; e += 256) bia[e] = rpb[e];
    __syncthreads();

    // ---- QKV GEMM: [64x192] @ [192x576]^T, 4x36 = 144 output tiles ----
    for (int t = wv; t < 144; t += 8) {
        int mt = t & 3, nt = t >> 2;
        // prefetch next tile's weight rows (L2-resident; hides hit latency)
        if (t + 8 < 144)
            __builtin_prefetch(qw + ((t + 8) >> 2) * 16 * CCH + ln * CCH, 0, 2);
        v8f acc = {};
#pragma unroll
        for (int ks = 0; ks < 6; ++ks) {
            v16bf a  = load_A(xo + (mt * 16) * CCH + ks * 32, CCH, ln, hi);
            v16bf bm = load_B(qw + (nt * 16) * CCH + ks * 32, CCH, ln, hi);
            acc = wmma_bf16(a, bm, acc);
        }
        int gc0 = nt * 16;                       // scalar: nt is wave-uniform
        if (gc0 < CCH) {                         // Q
#pragma unroll
            for (int r = 0; r < 8; ++r)
                qb[(mt * 16 + r + hi * 8) * CCH + gc0 + ln] = f2bf(acc[r]);
        } else if (gc0 < 2 * CCH) {              // K
#pragma unroll
            for (int r = 0; r < 8; ++r)
                kb[(mt * 16 + r + hi * 8) * CCH + gc0 - CCH + ln] = f2bf(acc[r]);
        } else {                                 // V stored transposed [dim][tok]
#pragma unroll
            for (int r = 0; r < 8; ++r)
                vT[(gc0 - 2 * CCH + ln) * TOK + (mt * 16 + r + hi * 8)] =
                    f2bf(acc[r]);
        }
    }
    __syncthreads();

    // ---- attention per head ----
    for (int h = 0; h < NHEAD; ++h) {
        // S = q @ k^T : 16 tiles, K = hd = 32 -> single WMMA each
#pragma unroll
        for (int tt = 0; tt < 2; ++tt) {
            int t  = wv + tt * 8;
            int mt = t >> 2, nt = t & 3;
            v16bf a  = load_A(qb + (mt * 16) * CCH + h * HD, CCH, ln, hi);
            v16bf bm = load_B(kb + (nt * 16) * CCH + h * HD, CCH, ln, hi);
            v8f acc = {};
            acc = wmma_bf16(a, bm, acc);
#pragma unroll
            for (int r = 0; r < 8; ++r) {
                int gm = mt * 16 + r + hi * 8;
                int gn = nt * 16 + ln;
                int idx = ((gm >> 3) - (gn >> 3) + 7) * 15 +
                          ((gm & 7)  - (gn & 7)  + 7);
                Sb[gm * TOK + gn] = acc[r] * scale + bia[idx * NHEAD + h];
            }
        }
        __syncthreads();

        // softmax over rows (64 rows, one thread each)
        if (tid < TOK) {
            float mx = -1e30f;
            for (int j = 0; j < TOK; ++j) mx = fmaxf(mx, Sb[tid * TOK + j]);
            float sum = 0.f;
            for (int j = 0; j < TOK; ++j) {
                float e = __expf(Sb[tid * TOK + j] - mx);
                Sb[tid * TOK + j] = e;
                sum += e;
            }
            float inv = 1.f / sum;
            for (int j = 0; j < TOK; ++j)
                Pb[tid * TOK + j] = f2bf(Sb[tid * TOK + j] * inv);
        }
        __syncthreads();

        // O_h = P @ V_h : 8 tiles (one per wave), K = 64 -> 2 WMMAs
        {
            int mt = wv >> 1, nt = wv & 1;
            v8f acc = {};
#pragma unroll
            for (int ks = 0; ks < 2; ++ks) {
                v16bf a  = load_A(Pb + (mt * 16) * TOK + ks * 32, TOK, ln, hi);
                v16bf bm = load_B(vT + (h * HD + nt * 16) * TOK + ks * 32,
                                  TOK, ln, hi);
                acc = wmma_bf16(a, bm, acc);
            }
#pragma unroll
            for (int r = 0; r < 8; ++r)
                xo[(mt * 16 + r + hi * 8) * CCH + h * HD + nt * 16 + ln] =
                    f2bf(acc[r]);
        }
        __syncthreads();
    }

    // ---- projection: O[64x192] @ proj_w^T + b, scatter to global ----
    float* ob = out + (size_t)bb * IMGW * IMGW * CCH;
    for (int t = wv; t < 48; t += 8) {
        int mt = t & 3, nt = t >> 2;
        if (t + 8 < 48)
            __builtin_prefetch(pw + ((t + 8) >> 2) * 16 * CCH + ln * CCH, 0, 2);
        v8f acc = {};
#pragma unroll
        for (int ks = 0; ks < 6; ++ks) {
            v16bf a  = load_A(xo + (mt * 16) * CCH + ks * 32, CCH, ln, hi);
            v16bf bm = load_B(pw + (nt * 16) * CCH + ks * 32, CCH, ln, hi);
            acc = wmma_bf16(a, bm, acc);
        }
        int gc = nt * 16 + ln;
        float pb = proj_b[gc];
#pragma unroll
        for (int r = 0; r < 8; ++r) {
            int gm  = mt * 16 + r + hi * 8;
            int row = wy * 8 + (gm >> 3), col = wx * 8 + (gm & 7);
            ob[((size_t)row * IMGW + col) * CCH + gc] = acc[r] + pb;
        }
    }
}

// ---------------------------------------------------------------------------
extern "C" void kernel_launch(void* const* d_in, const int* in_sizes, int n_in,
                              void* d_out, int out_size, void* d_ws, size_t ws_size,
                              hipStream_t stream) {
    const float* x      = (const float*)d_in[0];
    const float* qkv_w  = (const float*)d_in[1];
    const float* proj_w = (const float*)d_in[2];
    const float* proj_b = (const float*)d_in[3];
    const float* rpb    = (const float*)d_in[4];
    float* out = (float*)d_out;

    // workspace: bf16 weights
    __bf16* qw = (__bf16*)d_ws;                               // 576*192
    __bf16* pw = qw + 3 * CCH * CCH;                          // 192*192

    const int nB = in_sizes[0] / (IMGW * IMGW * CCH);         // batch (4)
    const int nWin = nB * WPX * WPX;                          // 4096

    cvt_weights_kernel<<<(3 * CCH * CCH + 255) / 256, 256, 0, stream>>>(
        qkv_w, proj_w, qw, pw);

    const size_t smem = (size_t)(4 * TOK * CCH) * sizeof(__bf16)  // x/O,Q,K,V^T
                      + (size_t)(TOK * TOK) * sizeof(float)       // S
                      + (size_t)(TOK * TOK) * sizeof(__bf16)      // P
                      + (size_t)(225 * NHEAD) * sizeof(float);    // bias
    swin_window_attn_kernel<<<nWin, 256, smem, stream>>>(
        x, qw, pw, proj_b, rpb, out);
    (void)n_in; (void)out_size; (void)ws_size;
}
